// GMAttend_31722628448373
// MI455X (gfx1250) — compile-verified, matrix operand order checked by reading, not run on hardware
//
#include <hip/hip_runtime.h>

// ---------------------------------------------------------------------------
// GMAttend fused pipeline for gfx1250 (MI455X), bf16 WMMA with f32 accumulate.
//
//   1) proj:    Q = X*Wq+bq, K = X*Wk+bk, V = X*Wv+bv   (bf16 out to ws)
//   2) qk:      L[g] = Q[g] * K[g]^T                     (bf16 logits to ws)
//   3) softmax: row-wise softmax of L in place           (bf16)
//   4) av:      out = gamma * (L[g] * V[g]) + x          (f32 to d_out)
//
// CDNA5 specifics: v_wmma_f32_16x16x32_bf16, ds_load_b128 fragment fills,
// global_load_async_to_lds_b128 staging (ASYNCcnt), double-buffered LDS.
// ---------------------------------------------------------------------------

typedef __attribute__((ext_vector_type(16))) __bf16          v16bf;
typedef __attribute__((ext_vector_type(8)))  unsigned        v8u;
typedef __attribute__((ext_vector_type(8)))  float           v8f;

#define BM   128
#define BN   128
#define BK   32
#define LSTR 40   // BK + 8 pad (ushorts): row stride = 80B = 16B-aligned rows

#define S_LEN 1024
#define H_LEN 1024
#define KD    128
#define NGRP  32

__device__ __forceinline__ unsigned short f2bf(float f) {
  unsigned u = __float_as_uint(f);
  return (unsigned short)((u + 0x7FFFu + ((u >> 16) & 1u)) >> 16);  // RNE
}
__device__ __forceinline__ float bf2f(unsigned short h) {
  return __uint_as_float(((unsigned)h) << 16);
}

// Async copy 16B global -> LDS (GLOBAL_LOAD_ASYNC_TO_LDS_B128, GV mode).
// LDS destination address = low 32 bits of the generic shared-memory address.
#define ASYNC_CP16(lds_ptr, g_ptr)                                          \
  asm volatile("global_load_async_to_lds_b128 %0, %1, off"                  \
               :: "v"((unsigned)(size_t)(lds_ptr)), "v"(g_ptr) : "memory")

__device__ __forceinline__ void async_wait0() {
  asm volatile("s_wait_asynccnt 0" ::: "memory");
}

// Load one 16x32 bf16 fragment from an LDS tile stored [16-major][K(=32)+pad].
// ISA 7.12.2 layout: lane<16 -> K {0..7,16..23}, lane>=16 -> K {8..15,24..31};
// each half is a contiguous 16B run -> two ds_load_b128.
__device__ __forceinline__ v16bf lds_frag(const unsigned short* rowbase, int lane) {
  const int r = lane & 15, half = lane >> 4;
  const unsigned short* p = rowbase + r * LSTR + half * 8;
  const uint4 lo = *(const uint4*)(p);        // K = half*8 + 0..7
  const uint4 hi = *(const uint4*)(p + 16);   // K = 16 + half*8 + 0..7
  v8u t;
  t[0] = lo.x; t[1] = lo.y; t[2] = lo.z; t[3] = lo.w;
  t[4] = hi.x; t[5] = hi.y; t[6] = hi.z; t[7] = hi.w;
  return __builtin_bit_cast(v16bf, t);
}

__device__ __forceinline__ v8f v8f_zero() {
  v8f z;
#pragma unroll
  for (int e = 0; e < 8; ++e) z[e] = 0.0f;
  return z;
}

__device__ __forceinline__ v8f wmma_bf16(v16bf a, v16bf b, v8f c) {
  return __builtin_amdgcn_wmma_f32_16x16x32_bf16(
      false, a, false, b, (short)0, c, false, false);
}

// ---------------------------------------------------------------------------
// Kernel 1: C(bf16)[M,N] = A(f32)[M,K] * W(f32)[K,N] + bias[N]
// 256 threads = 8 waves; block tile 128x128, k-chunk 32; double-buffered LDS.
// Wave w -> rows (w>>1)*32, cols (w&1)*64  (2x4 WMMA accumulators).
// ---------------------------------------------------------------------------
__global__ __launch_bounds__(256) void proj_kernel(
    const float* __restrict__ A, const float* __restrict__ W,
    const float* __restrict__ bias, unsigned short* __restrict__ C,
    int M, int N, int K) {
  __shared__ __align__(16) unsigned short As[2][BM * LSTR];
  __shared__ __align__(16) unsigned short Bs[2][BN * LSTR];
  const int t = threadIdx.x, lane = t & 31, wave = t >> 5;
  const int wm = wave >> 1, wn = wave & 1;
  const int row0 = blockIdx.y * BM, col0 = blockIdx.x * BN;

  v8f acc[2][4];
#pragma unroll
  for (int mi = 0; mi < 2; ++mi)
#pragma unroll
    for (int ni = 0; ni < 4; ++ni) acc[mi][ni] = v8f_zero();

  auto stage = [&](int kc, int buf) {
    // A: 128x32 f32 -> bf16, float4 reads, 8B LDS stores.
#pragma unroll
    for (int i = 0; i < 4; ++i) {
      int idx4 = t + i * 256;          // 1024 float4 per tile
      int r    = idx4 >> 3;
      int c4   = (idx4 & 7) * 4;
      float4 f = *(const float4*)(A + (size_t)(row0 + r) * K + kc + c4);
      unsigned short* d = &As[buf][r * LSTR + c4];
      d[0] = f2bf(f.x); d[1] = f2bf(f.y); d[2] = f2bf(f.z); d[3] = f2bf(f.w);
    }
    // W transposed: Bs[n][k] = W[(kc+k)*N + col0+n] (coalesced over n).
#pragma unroll
    for (int i = 0; i < 16; ++i) {
      int idx = t + i * 256;           // 4096 elements
      int k   = idx >> 7;
      int n   = idx & 127;
      Bs[buf][n * LSTR + k] = f2bf(W[(size_t)(kc + k) * N + col0 + n]);
    }
  };

  const int nc = K / BK;
  stage(0, 0);
  __syncthreads();
  for (int c = 0; c < nc; ++c) {
    const int cur = c & 1;
    if (c + 1 < nc) stage((c + 1) * BK, cur ^ 1);

    v16bf af[2], bfr[4];
#pragma unroll
    for (int mi = 0; mi < 2; ++mi)
      af[mi] = lds_frag(&As[cur][(wm * 32 + mi * 16) * LSTR], lane);
#pragma unroll
    for (int ni = 0; ni < 4; ++ni)
      bfr[ni] = lds_frag(&Bs[cur][(wn * 64 + ni * 16) * LSTR], lane);
#pragma unroll
    for (int mi = 0; mi < 2; ++mi)
#pragma unroll
      for (int ni = 0; ni < 4; ++ni)
        acc[mi][ni] = wmma_bf16(af[mi], bfr[ni], acc[mi][ni]);
    __syncthreads();
  }

  // Epilogue: +bias, store bf16.  C/D layout: lane&15 = N, lane>>4 -> M+8.
  const int nlo = lane & 15, mb = (lane >> 4) * 8;
#pragma unroll
  for (int mi = 0; mi < 2; ++mi)
#pragma unroll
    for (int ni = 0; ni < 4; ++ni) {
      int col = col0 + wn * 64 + ni * 16 + nlo;
      int row = row0 + wm * 32 + mi * 16 + mb;
      float b = bias[col];
#pragma unroll
      for (int j = 0; j < 8; ++j)
        C[(size_t)(row + j) * N + col] = f2bf(acc[mi][ni][j] + b);
    }
}

// ---------------------------------------------------------------------------
// Kernel 2: per-group logits  L[g] = Q[g] * K[g]^T   (reduction dim = KD)
// Q and K row-major [S,KD] bf16: both tiles stage identically via async B128.
// ---------------------------------------------------------------------------
__global__ __launch_bounds__(256) void qk_kernel(
    const unsigned short* __restrict__ Q, const unsigned short* __restrict__ Km,
    unsigned short* __restrict__ L) {
  __shared__ __align__(16) unsigned short As[2][BM * LSTR];
  __shared__ __align__(16) unsigned short Bs[2][BN * LSTR];
  const int t = threadIdx.x, lane = t & 31, wave = t >> 5;
  const int wm = wave >> 1, wn = wave & 1;
  const int g = blockIdx.z;
  const unsigned short* Qg = Q  + (size_t)g * S_LEN * KD;
  const unsigned short* Kg = Km + (size_t)g * S_LEN * KD;
  unsigned short*       Lg = L  + (size_t)g * S_LEN * S_LEN;
  const int row0 = blockIdx.y * BM, col0 = blockIdx.x * BN;

  v8f acc[2][4];
#pragma unroll
  for (int mi = 0; mi < 2; ++mi)
#pragma unroll
    for (int ni = 0; ni < 4; ++ni) acc[mi][ni] = v8f_zero();

  auto stage = [&](int kc, int buf) {
#pragma unroll
    for (int i = 0; i < 2; ++i) {
      int idx = t + i * 256;           // 512 x 16B per tile
      int r   = idx >> 2;
      int c8  = (idx & 3) * 8;
      ASYNC_CP16(&As[buf][r * LSTR + c8], Qg + (size_t)(row0 + r) * KD + kc + c8);
      ASYNC_CP16(&Bs[buf][r * LSTR + c8], Kg + (size_t)(col0 + r) * KD + kc + c8);
    }
  };

  const int nc = KD / BK;              // 4 chunks
  stage(0, 0);
  async_wait0();
  __syncthreads();
  for (int c = 0; c < nc; ++c) {
    const int cur = c & 1;
    if (c + 1 < nc) stage((c + 1) * BK, cur ^ 1);

    v16bf af[2], bfr[4];
#pragma unroll
    for (int mi = 0; mi < 2; ++mi)
      af[mi] = lds_frag(&As[cur][(wm * 32 + mi * 16) * LSTR], lane);
#pragma unroll
    for (int ni = 0; ni < 4; ++ni)
      bfr[ni] = lds_frag(&Bs[cur][(wn * 64 + ni * 16) * LSTR], lane);
#pragma unroll
    for (int mi = 0; mi < 2; ++mi)
#pragma unroll
      for (int ni = 0; ni < 4; ++ni)
        acc[mi][ni] = wmma_bf16(af[mi], bfr[ni], acc[mi][ni]);

    async_wait0();
    __syncthreads();
  }

  const int nlo = lane & 15, mb = (lane >> 4) * 8;
#pragma unroll
  for (int mi = 0; mi < 2; ++mi)
#pragma unroll
    for (int ni = 0; ni < 4; ++ni) {
      int col = col0 + wn * 64 + ni * 16 + nlo;
      int row = row0 + wm * 32 + mi * 16 + mb;
#pragma unroll
      for (int j = 0; j < 8; ++j)
        Lg[(size_t)(row + j) * S_LEN + col] = f2bf(acc[mi][ni][j]);
    }
}

// ---------------------------------------------------------------------------
// Kernel 3: in-place row softmax over 1024 bf16 logits; one wave32 per row.
// ---------------------------------------------------------------------------
__global__ __launch_bounds__(256) void softmax_kernel(unsigned short* __restrict__ L) {
  const int wave = threadIdx.x >> 5, lane = threadIdx.x & 31;
  const size_t row = (size_t)blockIdx.x * 8 + wave;   // 32768 rows total
  unsigned short* p = L + row * S_LEN;

  float v[32];
  float m = -3.4e38f;
#pragma unroll
  for (int j = 0; j < 32; ++j) {
    v[j] = bf2f(p[lane + j * 32]);
    m = fmaxf(m, v[j]);
  }
#pragma unroll
  for (int s = 16; s > 0; s >>= 1) m = fmaxf(m, __shfl_xor(m, s, 32));
  float sum = 0.0f;
#pragma unroll
  for (int j = 0; j < 32; ++j) { v[j] = __expf(v[j] - m); sum += v[j]; }
#pragma unroll
  for (int s = 16; s > 0; s >>= 1) sum += __shfl_xor(sum, s, 32);
  const float inv = 1.0f / sum;
#pragma unroll
  for (int j = 0; j < 32; ++j) p[lane + j * 32] = f2bf(v[j] * inv);
}

// ---------------------------------------------------------------------------
// Kernel 4: out = gamma * (Att[g] * V[g]) + x    (f32 output)
// A (att rows) staged async; V staged transposed via VALU gather.
// ---------------------------------------------------------------------------
__global__ __launch_bounds__(256) void av_kernel(
    const unsigned short* __restrict__ Att, const unsigned short* __restrict__ V,
    const float* __restrict__ x, const float* __restrict__ gamma,
    float* __restrict__ out) {
  __shared__ __align__(16) unsigned short As[2][BM * LSTR];
  __shared__ __align__(16) unsigned short Bs[2][BN * LSTR];
  const int t = threadIdx.x, lane = t & 31, wave = t >> 5;
  const int wm = wave >> 1, wn = wave & 1;
  const int g = blockIdx.z;
  const unsigned short* Ag = Att + (size_t)g * S_LEN * S_LEN;
  const unsigned short* Vg = V   + (size_t)g * S_LEN * H_LEN;
  const int row0 = blockIdx.y * BM, col0 = blockIdx.x * BN;

  v8f acc[2][4];
#pragma unroll
  for (int mi = 0; mi < 2; ++mi)
#pragma unroll
    for (int ni = 0; ni < 4; ++ni) acc[mi][ni] = v8f_zero();

  auto stage = [&](int kc, int buf) {
    // Att rows: k contiguous -> async B128 copies.
#pragma unroll
    for (int i = 0; i < 2; ++i) {
      int idx = t + i * 256;
      int r   = idx >> 2;
      int c8  = (idx & 3) * 8;
      ASYNC_CP16(&As[buf][r * LSTR + c8], Ag + (size_t)(row0 + r) * S_LEN + kc + c8);
    }
    // V transposed: Bs[n][k] = Vg[(kc+k)*H + col0+n] (coalesced over n).
#pragma unroll
    for (int i = 0; i < 16; ++i) {
      int idx = t + i * 256;
      int k   = idx >> 7;
      int n   = idx & 127;
      Bs[buf][n * LSTR + k] = Vg[(size_t)(kc + k) * H_LEN + col0 + n];
    }
  };

  const int nc = S_LEN / BK;           // 32 chunks
  stage(0, 0);
  async_wait0();
  __syncthreads();
  for (int c = 0; c < nc; ++c) {
    const int cur = c & 1;
    if (c + 1 < nc) stage((c + 1) * BK, cur ^ 1);

    v16bf af[2], bfr[4];
#pragma unroll
    for (int mi = 0; mi < 2; ++mi)
      af[mi] = lds_frag(&As[cur][(wm * 32 + mi * 16) * LSTR], lane);
#pragma unroll
    for (int ni = 0; ni < 4; ++ni)
      bfr[ni] = lds_frag(&Bs[cur][(wn * 64 + ni * 16) * LSTR], lane);
#pragma unroll
    for (int mi = 0; mi < 2; ++mi)
#pragma unroll
      for (int ni = 0; ni < 4; ++ni)
        acc[mi][ni] = wmma_bf16(af[mi], bfr[ni], acc[mi][ni]);

    async_wait0();
    __syncthreads();
  }

  const float gm = gamma[0];
  const int nlo = lane & 15, mb = (lane >> 4) * 8;
#pragma unroll
  for (int mi = 0; mi < 2; ++mi)
#pragma unroll
    for (int ni = 0; ni < 4; ++ni) {
      int col = col0 + wn * 64 + ni * 16 + nlo;
      int row = row0 + wm * 32 + mi * 16 + mb;
#pragma unroll
      for (int j = 0; j < 8; ++j) {
        size_t o = ((size_t)g * S_LEN + row + j) * H_LEN + col;
        out[o] = gm * acc[mi][ni][j] + x[o];
      }
    }
}

// ---------------------------------------------------------------------------
extern "C" void kernel_launch(void* const* d_in, const int* in_sizes, int n_in,
                              void* d_out, int out_size, void* d_ws, size_t ws_size,
                              hipStream_t stream) {
  (void)in_sizes; (void)n_in; (void)out_size; (void)ws_size;
  const float* x     = (const float*)d_in[0];
  const float* Wq    = (const float*)d_in[1];
  const float* bq    = (const float*)d_in[2];
  const float* Wk    = (const float*)d_in[3];
  const float* bk    = (const float*)d_in[4];
  const float* Wv    = (const float*)d_in[5];
  const float* bv    = (const float*)d_in[6];
  const float* gamma = (const float*)d_in[7];
  float* out = (float*)d_out;

  const size_t M = (size_t)NGRP * S_LEN;           // 32768 flattened rows
  unsigned short* Qw = (unsigned short*)d_ws;      // [M, KD]
  unsigned short* Kw = Qw + M * KD;                // [M, KD]
  unsigned short* Vw = Kw + M * KD;                // [M, H]
  unsigned short* Lw = Vw + M * H_LEN;             // [M, S]

  proj_kernel<<<dim3(KD / BN,    M / BM), 256, 0, stream>>>(x, Wq, bq, Qw, (int)M, KD,    H_LEN);
  proj_kernel<<<dim3(KD / BN,    M / BM), 256, 0, stream>>>(x, Wk, bk, Kw, (int)M, KD,    H_LEN);
  proj_kernel<<<dim3(H_LEN / BN, M / BM), 256, 0, stream>>>(x, Wv, bv, Vw, (int)M, H_LEN, H_LEN);
  qk_kernel<<<dim3(S_LEN / BN, S_LEN / BM, NGRP), 256, 0, stream>>>(Qw, Kw, Lw);
  softmax_kernel<<<(unsigned)(M / 8), 256, 0, stream>>>(Lw);
  av_kernel<<<dim3(H_LEN / BN, S_LEN / BM, NGRP), 256, 0, stream>>>(Lw, Vw, x, gamma, out);
}
